// IRCA_21345987461465
// MI455X (gfx1250) — compile-verified, matrix operand order checked by reading, not run on hardware
//
#include <hip/hip_runtime.h>
#include <math.h>

typedef __attribute__((ext_vector_type(16))) __bf16 v16bf;
typedef __attribute__((ext_vector_type(8)))  float  v8f;

#define Bn 16
#define Cn 256
#define HWn 4096
#define NPAD 272        // 17 tiles of 16 (256 ctx + 4 gates + 12 zero pad)
#define ROWU 40         // LDS row stride in ushorts (80 B: 16B-aligned, banks 20r%64 distinct)

// ---- CDNA5 async global->LDS copy (16 B per lane), ASYNCcnt-tracked
__device__ __forceinline__ void async_copy16(unsigned lds_off, const void* g) {
  asm volatile("global_load_async_to_lds_b128 %0, %1, off"
               :: "v"(lds_off), "v"(g) : "memory");
}
__device__ __forceinline__ void wait_async0() {
  asm volatile("s_wait_asynccnt 0x0" ::: "memory");
}
__device__ __forceinline__ unsigned lds_off(const void* p) {
  return (unsigned)(uintptr_t)p;    // LDS aperture: low 32 bits = LDS byte offset
}

// ---------------- K0a: convert conv_w (260x256 f32) -> bf16, padded to 272 rows
__global__ void k_wconv(const float* __restrict__ w, __bf16* __restrict__ wb) {
  int i = blockIdx.x * blockDim.x + threadIdx.x;   // 272*256
  int o = i >> 8;
  float v = (o < Cn + 4) ? w[i] : 0.0f;
  wb[i] = (__bf16)v;
}

// ---------------- K0b: transpose+downcast x -> pixel-major bf16
// out layout: [b][pblk(32)][pixel(128)][channel(256)] bf16 (channel contiguous)
__global__ void __launch_bounds__(256)
k_xt(const float* __restrict__ x, unsigned* __restrict__ xt) {
  __shared__ float tile[64][129];                  // pad kills bank conflicts
  int blk = blockIdx.x;                            // b(16) * cblk(4) * pblk(32)
  int b = blk >> 7, cblk = (blk >> 5) & 3, pblk = blk & 31;
  const float* xp = x + (size_t)(b * Cn + cblk * 64) * HWn + pblk * 128;
  for (int i = threadIdx.x; i < 64 * 128; i += 256) {
    int c = i >> 7, p = i & 127;
    tile[c][p] = xp[(size_t)c * HWn + p];          // coalesced over p
  }
  __syncthreads();
  unsigned* op = xt + (size_t)(b * 32 + pblk) * 128 * (Cn / 2) + cblk * 32;
  for (int i = threadIdx.x; i < 4096; i += 256) {  // 128 px * 32 ch-pairs
    int p = i >> 5, cp = i & 31;
    union { __bf16 h[2]; unsigned u; } v;
    v.h[0] = (__bf16)tile[2 * cp][p];
    v.h[1] = (__bf16)tile[2 * cp + 1][p];
    op[(size_t)p * (Cn / 2) + cp] = v.u;           // coalesced over cp
  }
}

// ---------------- K1: conv1x1 bf16 WMMA GEMM.
// All LDS staging via async-to-LDS, double-buffered; B fragments 4-deep pipelined.
// amdgpu_waves_per_eu(1) relaxes the VGPR budget so the 4 B-buffers stay live.
// Block = 256 threads = 8 waves; block owns 128 pixels; wave owns 16 px x 272 outs.
__global__ void __launch_bounds__(256) __attribute__((amdgpu_waves_per_eu(1)))
k_gemm(const unsigned short* __restrict__ xt16, const unsigned short* __restrict__ wg16,
       const float* __restrict__ cb, float* __restrict__ ctx,
       float* __restrict__ gates) {
  __shared__ __align__(16) unsigned short Ab[2][128 * ROWU];   // 2 x 10,240 B
  __shared__ __align__(16) unsigned short Bb[2][NPAD * ROWU];  // 2 x 21,760 B  (total 64,000 B)

  int tid = threadIdx.x;
  int wv = tid >> 5, lane = tid & 31;
  int blk = blockIdx.x;               // 512 blocks
  int b  = blk >> 5;
  int pb = blk & 31;
  const unsigned short* xg = xt16 + (size_t)(b * 32 + pb) * 128 * Cn;  // pixel-major rows

  int r  = lane & 15;                 // fragment row (pixel / out-channel)
  int kh = (lane >> 4) << 4;          // K half (ushort offset 0 or 16)

  v8f acc[17];
#pragma unroll
  for (int t = 0; t < 17; ++t) acc[t] = (v8f){0,0,0,0,0,0,0,0};

  // async-stage one K-chunk (32 channels) into buffer `buf`
  auto stage = [&](int kc, int buf) {
    for (int i = tid; i < 512; i += 256) {         // A: 128 px * 4 x 16B
      int p = i >> 2, j = i & 3;
      async_copy16(lds_off(Ab[buf] + p * ROWU) + j * 16,
                   xg + (size_t)p * Cn + kc + j * 8);
    }
    for (int i = tid; i < 1088; i += 256) {        // B: 272 rows * 4 x 16B
      int o = i >> 2, j = i & 3;
      async_copy16(lds_off(Bb[buf] + o * ROWU) + j * 16,
                   wg16 + (size_t)o * Cn + kc + j * 8);
    }
  };

  stage(0, 0);
  wait_async0();
  __syncthreads();

  for (int c = 0; c < 8; ++c) {       // 8 K-chunks of 32
    int cur = c & 1;
    if (c < 7) stage((c + 1) * 32, cur ^ 1);   // fire-and-forget next chunk

    // ---- compute: 17 N-tiles, B fragments 4-deep register pipeline
    const unsigned short* Abuf = Ab[cur];
    const unsigned short* Bbuf = Bb[cur];
    union U16 { uint4 q[2]; v16bf v; };
    U16 au, bu[4];
    const uint4* ap = (const uint4*)(Abuf + (wv * 16 + r) * ROWU + kh);
    au.q[0] = ap[0]; au.q[1] = ap[1];
#pragma unroll
    for (int t = 0; t < 3; ++t) {
      const uint4* bp = (const uint4*)(Bbuf + (t * 16 + r) * ROWU + kh);
      bu[t].q[0] = bp[0]; bu[t].q[1] = bp[1];
    }
#pragma unroll
    for (int t = 0; t < 17; ++t) {
      if (t + 3 < 17) {
        const uint4* bp = (const uint4*)(Bbuf + ((t + 3) * 16 + r) * ROWU + kh);
        bu[(t + 3) & 3].q[0] = bp[0];
        bu[(t + 3) & 3].q[1] = bp[1];
      }
      acc[t] = __builtin_amdgcn_wmma_f32_16x16x32_bf16(
          false, au.v, false, bu[t & 3].v, (short)0, acc[t], false, false);
    }

    wait_async0();                    // next chunk landed in LDS
    __syncthreads();                  // all waves' staging visible; WAR-safe
  }

  // ---- store: VGPR j -> pixel (j + 8*(lane>=16)), channel = tile*16 + lane%16
  int P0 = pb << 7;
  int p0 = P0 + wv * 16;
  int nn = lane & 15;
  int mo = (lane >> 4) << 3;
#pragma unroll
  for (int t = 0; t < 16; ++t) {
    int o = t * 16 + nn;
    float bias = cb[o];
    float* dp = ctx + ((size_t)(b * Cn + o) * HWn + p0 + mo);
#pragma unroll
    for (int j = 0; j < 8; ++j) dp[j] = acc[t][j] + bias;
  }
  if (nn < 4) {                       // gate channels 256..259
    float bias = cb[Cn + nn];
    float* dp = gates + ((size_t)(b * 4 + nn) * HWn + p0 + mo);
#pragma unroll
    for (int j = 0; j < 8; ++j) dp[j] = acc[16][j] + bias;
  }
}

// ---------------- K3: depthwise conv (SAME, zero pad) + exact GELU + gated acc
template <int R>
__global__ void __launch_bounds__(256)
k_dwconv(const float* __restrict__ in, const float* __restrict__ fw,
         const float* __restrict__ gates, int gidx,
         float* __restrict__ outctx, float* __restrict__ ctx_all, int first) {
  const int S = 64 + 2 * R;
  __shared__ float tile[(64 + 2 * R) * (64 + 2 * R)];
  int bc = blockIdx.x, b = bc >> 8, c = bc & 255;
  const float* ip = in + (size_t)bc * HWn;
  for (int i = threadIdx.x; i < S * S; i += blockDim.x) {
    int y = i / S - R, xx = i % S - R;
    float v = 0.0f;
    if (y >= 0 && y < 64 && xx >= 0 && xx < 64) v = ip[y * 64 + xx];
    tile[i] = v;
  }
  float wk[(2 * R + 1) * (2 * R + 1)];
#pragma unroll
  for (int i = 0; i < (2 * R + 1) * (2 * R + 1); ++i)
    wk[i] = fw[c * (2 * R + 1) * (2 * R + 1) + i];
  __syncthreads();

  const float* gp = gates + (size_t)(b * 4 + gidx) * HWn;
  float* op = outctx + (size_t)bc * HWn;
  float* ap = ctx_all + (size_t)bc * HWn;
  for (int p = threadIdx.x; p < HWn; p += blockDim.x) {
    int y = p >> 6, xx = p & 63;
    float s = 0.0f;
#pragma unroll
    for (int dy = 0; dy < 2 * R + 1; ++dy)
#pragma unroll
      for (int dx = 0; dx < 2 * R + 1; ++dx)
        s = fmaf(wk[dy * (2 * R + 1) + dx], tile[(y + dy) * S + (xx + dx)], s);
    float g = 0.5f * s * (1.0f + erff(s * 0.70710678118f));   // exact GELU
    op[p] = g;
    float add = g * gp[p];
    ap[p] = first ? add : (ap[p] + add);
  }
}

// ---------------- K4a: per-(b,c) spatial mean of final ctx
__global__ void __launch_bounds__(256) k_mean(const float* __restrict__ ctx,
                                              float* __restrict__ cg) {
  __shared__ float red[256];
  const float* p = ctx + (size_t)blockIdx.x * HWn;
  float s = 0.0f;
  for (int i = threadIdx.x; i < HWn; i += 256) s += p[i];
  red[threadIdx.x] = s; __syncthreads();
  for (int st = 128; st > 0; st >>= 1) {
    if (threadIdx.x < st) red[threadIdx.x] += red[threadIdx.x + st];
    __syncthreads();
  }
  if (threadIdx.x == 0) cg[blockIdx.x] = red[0] * (1.0f / HWn);
}

// ---------------- K4b: ctx_all += ctx_global * gate3 (per pixel)
__global__ void k_addglobal(float* __restrict__ ca, const float* __restrict__ cg,
                            const float* __restrict__ gates) {
  size_t i = (size_t)blockIdx.x * blockDim.x + threadIdx.x;   // B*C*HW
  int p = (int)(i & (HWn - 1));
  int bc = (int)(i >> 12);
  int b = bc >> 8;
  ca[i] += cg[bc] * gates[(size_t)(b * 4 + 3) * HWn + p];
}

// ---------------- K5a: key logits k[b,p] = key_w . ctx_all[b,:,p] + key_b
__global__ void k_key(const float* __restrict__ ca, const float* __restrict__ kw,
                      const float* __restrict__ kb, float* __restrict__ klog) {
  int i = blockIdx.x * blockDim.x + threadIdx.x;   // B*HW
  int b = i >> 12, p = i & (HWn - 1);
  const float* q = ca + (size_t)b * Cn * HWn + p;
  float s = kb[0];
  for (int c = 0; c < Cn; ++c) s = fmaf(q[(size_t)c * HWn], kw[c], s);
  klog[i] = s;
}

// ---------------- K5b: softmax over HW per batch (in place)
__global__ void __launch_bounds__(256) k_softmax(float* __restrict__ klog) {
  __shared__ float red[256];
  float* p = klog + (size_t)blockIdx.x * HWn;
  int tid = threadIdx.x;
  float m = -1e30f;
  for (int i = tid; i < HWn; i += 256) m = fmaxf(m, p[i]);
  red[tid] = m; __syncthreads();
  for (int st = 128; st > 0; st >>= 1) {
    if (tid < st) red[tid] = fmaxf(red[tid], red[tid + st]);
    __syncthreads();
  }
  float mx = red[0]; __syncthreads();
  float s = 0.0f;
  for (int i = tid; i < HWn; i += 256) s += expf(p[i] - mx);
  red[tid] = s; __syncthreads();
  for (int st = 128; st > 0; st >>= 1) {
    if (tid < st) red[tid] += red[tid + st];
    __syncthreads();
  }
  float inv = 1.0f / red[0];
  for (int i = tid; i < HWn; i += 256) p[i] = expf(p[i] - mx) * inv;
}

// ---------------- K5c: qk[b,c] = sum_p ctx_all[b,c,p] * k[b,p]
__global__ void __launch_bounds__(256) k_qk(const float* __restrict__ ca,
                                            const float* __restrict__ k,
                                            float* __restrict__ qk) {
  __shared__ float red[256];
  int bc = blockIdx.x, b = bc >> 8;
  const float* q = ca + (size_t)bc * HWn;
  const float* kp = k + (size_t)b * HWn;
  float s = 0.0f;
  for (int i = threadIdx.x; i < HWn; i += 256) s = fmaf(q[i], kp[i], s);
  red[threadIdx.x] = s; __syncthreads();
  for (int st = 128; st > 0; st >>= 1) {
    if (threadIdx.x < st) red[threadIdx.x] += red[threadIdx.x + st];
    __syncthreads();
  }
  if (threadIdx.x == 0) qk[bc] = red[0];
}

// ---------------- K6: tiny MLP per batch: v1 -> LN -> ReLU -> v2
__global__ void __launch_bounds__(256)
k_mlp(const float* __restrict__ qk, const float* __restrict__ v1w,
      const float* __restrict__ v1b, const float* __restrict__ lnw,
      const float* __restrict__ lnb, const float* __restrict__ v2w,
      const float* __restrict__ v2b, float* __restrict__ vout) {
  __shared__ float red[256];
  __shared__ float v16[16];
  __shared__ float stats[2];
  int b = blockIdx.x, tid = threadIdx.x;
  int oc = tid >> 4, ch = tid & 15;
  float s = 0.0f;
  for (int i = 0; i < 16; ++i)
    s = fmaf(v1w[oc * Cn + ch * 16 + i], qk[b * Cn + ch * 16 + i], s);
  red[tid] = s; __syncthreads();
  if (ch == 0) {
    float t = 0.0f;
    for (int i = 0; i < 16; ++i) t += red[oc * 16 + i];
    v16[oc] = t + v1b[oc];
  }
  __syncthreads();
  if (tid == 0) {
    float mu = 0.0f, sq = 0.0f;
    for (int i = 0; i < 16; ++i) { mu += v16[i]; sq += v16[i] * v16[i]; }
    mu *= (1.0f / 16.0f);
    stats[0] = mu;
    stats[1] = sq * (1.0f / 16.0f) - mu * mu;
  }
  __syncthreads();
  if (tid < 16) {
    float z = (v16[tid] - stats[0]) * rsqrtf(stats[1] + 1e-5f);
    z = z * lnw[tid] + lnb[tid];
    v16[tid] = fmaxf(z, 0.0f);
  }
  __syncthreads();
  float o = v2b[tid];
  for (int i = 0; i < 16; ++i) o = fmaf(v2w[tid * 16 + i], v16[i], o);
  vout[b * Cn + tid] = o;
}

// ---------------- K7: out += broadcast v
__global__ void k_addv(float* __restrict__ out, const float* __restrict__ vout) {
  size_t i = (size_t)blockIdx.x * blockDim.x + threadIdx.x;
  int bc = (int)(i >> 12);
  out[i] += vout[bc];
}

extern "C" void kernel_launch(void* const* d_in, const int* in_sizes, int n_in,
                              void* d_out, int out_size, void* d_ws, size_t ws_size,
                              hipStream_t stream) {
  const float* x      = (const float*)d_in[0];
  const float* conv_w = (const float*)d_in[1];
  const float* conv_b = (const float*)d_in[2];
  const float* fw3    = (const float*)d_in[3];
  const float* fw5    = (const float*)d_in[4];
  const float* fw7    = (const float*)d_in[5];
  const float* key_w  = (const float*)d_in[6];
  const float* key_b  = (const float*)d_in[7];
  const float* v1_w   = (const float*)d_in[8];
  const float* v1_b   = (const float*)d_in[9];
  const float* ln_w   = (const float*)d_in[10];
  const float* ln_b   = (const float*)d_in[11];
  const float* v2_w   = (const float*)d_in[12];
  const float* v2_b   = (const float*)d_in[13];
  float* out = (float*)d_out;

  char* ws = (char*)d_ws;
  size_t off = 0;
  auto alloc = [&](size_t bytes) {
    char* p = ws + off;
    off += (bytes + 255) & ~(size_t)255;
    return p;
  };
  __bf16* wbf  = (__bf16*)alloc((size_t)NPAD * Cn * 2);
  __bf16* xbf  = (__bf16*)alloc((size_t)Bn * Cn * HWn * 2);   // pixel-major transposed
  float* ctxA  = (float*)alloc((size_t)Bn * Cn * HWn * 4);
  float* ctxB  = (float*)alloc((size_t)Bn * Cn * HWn * 4);
  float* gates = (float*)alloc((size_t)Bn * 4 * HWn * 4);
  float* cg    = (float*)alloc((size_t)Bn * Cn * 4);
  float* klog  = (float*)alloc((size_t)Bn * HWn * 4);
  float* qkb   = (float*)alloc((size_t)Bn * Cn * 4);
  float* vob   = (float*)alloc((size_t)Bn * Cn * 4);

  // One-time downcast / transpose passes
  k_wconv<<<NPAD, 256, 0, stream>>>(conv_w, wbf);
  k_xt<<<Bn * 4 * 32, 256, 0, stream>>>(x, (unsigned*)xbf);

  // 1x1 conv GEMM (bf16 WMMA, async-LDS double-buffered): ctxA + gates
  k_gemm<<<512, 256, 0, stream>>>((const unsigned short*)xbf,
                                  (const unsigned short*)wbf,
                                  conv_b, ctxA, gates);

  // Gated multi-scale depthwise stack; d_out accumulates ctx_all
  k_dwconv<1><<<Bn * Cn, 256, 0, stream>>>(ctxA, fw3, gates, 0, ctxB, out, 1);
  k_dwconv<2><<<Bn * Cn, 256, 0, stream>>>(ctxB, fw5, gates, 1, ctxA, out, 0);
  k_dwconv<3><<<Bn * Cn, 256, 0, stream>>>(ctxA, fw7, gates, 2, ctxB, out, 0);

  // Global context gate
  k_mean<<<Bn * Cn, 256, 0, stream>>>(ctxB, cg);
  k_addglobal<<<(Bn * Cn * HWn) / 256, 256, 0, stream>>>(out, cg, gates);

  // Attention pooling
  k_key<<<(Bn * HWn) / 256, 256, 0, stream>>>(out, key_w, key_b, klog);
  k_softmax<<<Bn, 256, 0, stream>>>(klog);
  k_qk<<<Bn * Cn, 256, 0, stream>>>(out, klog, qkb);

  // Per-batch MLP + broadcast add
  k_mlp<<<Bn, 256, 0, stream>>>(qkb, v1_w, v1_b, ln_w, ln_b, v2_w, v2_b, vob);
  k_addv<<<(Bn * Cn * HWn) / 256, 256, 0, stream>>>(out, vob);
}
